// GeometricMultiHeadAttention_30820685316462
// MI455X (gfx1250) — matrix operand order repaired and checked by reading, not compiled, not run on hardware
//
#include <hip/hip_runtime.h>

// ---------------------------------------------------------------------------
// GeometricMultiHeadAttention for MI455X (gfx1250, wave32, WMMA bf16 + TDM)
//
// Roofline: fused flash attention avoids 128MB logit round-trips; the staging
// working set (~90MB) lives in the 192MB L2. GEMMs use bf16-in/f32-acc WMMA
// (16x16x32) — the fp32 WMMA shape (16x16x4) has 8x less K per instruction.
// GEMM tiles are staged into LDS by the Tensor Data Mover (tensor_load_to_lds,
// double-buffered, s_wait_tensorcnt) so fragment reads hit LDS and the DMA
// overlaps WMMA compute.
// ---------------------------------------------------------------------------

#define B_  2
#define L_  1024
#define D_  1024
#define H_  16
#define HD_ 64

typedef __bf16 v16bf __attribute__((ext_vector_type(16)));
typedef __bf16 v8bf  __attribute__((ext_vector_type(8)));
typedef float  v8f   __attribute__((ext_vector_type(8)));
typedef unsigned int v4u __attribute__((ext_vector_type(4)));
typedef int v4i __attribute__((ext_vector_type(4)));
typedef int v8i __attribute__((ext_vector_type(8)));

#ifndef __has_builtin
#define __has_builtin(x) 0
#endif
#if __has_builtin(__builtin_amdgcn_tensor_load_to_lds) && \
    __has_builtin(__builtin_amdgcn_s_wait_tensorcnt)
#define HAVE_TDM 1
#else
#define HAVE_TDM 0
#endif

__device__ __forceinline__ v8f vzero8() {
  v8f r;
#pragma unroll
  for (int i = 0; i < 8; ++i) r[i] = 0.f;
  return r;
}

// A fragment: 16x32 bf16, row-major source with leading dim ld (elements).
// Lanes 0-15: row M=lane, K=[0..15]; lanes 16-31: K=[16..31].
__device__ __forceinline__ v16bf load_a_frag(const __bf16* base, int ld, int lane) {
  int m = lane & 15;
  int koff = (lane >> 4) * 16;
  const __bf16* p = base + (size_t)m * ld + koff;
  v8bf lo = *(const v8bf*)p;
  v8bf hi = *(const v8bf*)(p + 8);
  v16bf r;
#pragma unroll
  for (int i = 0; i < 8; ++i) { r[i] = lo[i]; r[i + 8] = hi[i]; }
  return r;
}

// B fragment: 32x16 bf16 (KxN) where source is stored N-major (row n holds K
// contiguously with leading dim ld). Lanes 0-15: K=[0..15]; lanes 16-31: K=[16..31].
__device__ __forceinline__ v16bf load_b_frag(const __bf16* base, int ld, int lane) {
  int n = lane & 15;
  int koff = (lane >> 4) * 16;
  return *(const v16bf*)(base + (size_t)n * ld + koff);
}

__device__ __forceinline__ v8f wmma_bf16(v16bf a, v16bf b, v8f c) {
  return __builtin_amdgcn_wmma_f32_16x16x32_bf16(false, a, false, b, (short)0, c,
                                                 false, false);
}

__device__ __forceinline__ float rope_freq(int i /* 0..31 */) {
  // 2*pi / 10000^(i/31)
  return 6.283185307179586f * __expf(-(float)i * (9.210340371976184f / 31.f));
}

#if HAVE_TDM
// Issue a TDM 2D tile load: tile_h x tile_w bf16 elements from a row-major
// tensor (leading dim stride0 elements) into LDS at lds_byte_off.
// D# layout per CDNA5 ISA (group0 128b, group1 256b).
__device__ __forceinline__ void tdm_load_2d(const void* gaddr, unsigned lds_byte_off,
                                            unsigned tensor_w, unsigned tensor_h,
                                            unsigned tile_w, unsigned tile_h,
                                            unsigned stride0) {
  unsigned long long ga = (unsigned long long)gaddr;
  v4u g0;
  g0.x = 1u;                                   // count=1 valid user descriptor
  g0.y = lds_byte_off;                         // lds_addr [63:32]
  g0.z = (unsigned)ga;                         // global_addr [95:64]
  g0.w = (unsigned)(ga >> 32) | (2u << 30);    // global_addr hi + type=2
  v8i g1;
  g1[0] = (int)(1u << 16);                     // data_size=1 (2 bytes/elem)
  g1[1] = (int)((tensor_w & 0xFFFFu) << 16);   // tensor_dim0[15:0] @ bits 63:48
  g1[2] = (int)((tensor_w >> 16) | ((tensor_h & 0xFFFFu) << 16));
  g1[3] = (int)((tensor_h >> 16) | (tile_w << 16));   // tile_dim0 @ 127:112
  g1[4] = (int)(tile_h & 0xFFFFu);             // tile_dim1 (tile_dim2=0)
  g1[5] = (int)stride0;                        // tensor_dim0_stride[31:0]
  g1[6] = 0;
  g1[7] = 0;
  v4i zz;
#pragma unroll
  for (int i = 0; i < 4; ++i) zz[i] = 0;
#if __clang_major__ >= 23
  v8i z8;
#pragma unroll
  for (int i = 0; i < 8; ++i) z8[i] = 0;
  __builtin_amdgcn_tensor_load_to_lds(g0, g1, zz, zz, z8, 0);
#else
  __builtin_amdgcn_tensor_load_to_lds(g0, g1, zz, zz, 0);
#endif
}
#endif  // HAVE_TDM

// ---------------------------------------------------------------------------
// fp32 -> bf16 conversion
// ---------------------------------------------------------------------------
__global__ void __launch_bounds__(256) cvt_bf16_kernel(const float* __restrict__ src,
                                                       __bf16* __restrict__ dst, int n) {
  int i = blockIdx.x * 256 + threadIdx.x;
  if (i < n) dst[i] = (__bf16)src[i];
}

// ---------------------------------------------------------------------------
// C = A(M,K) @ Bt(N,K)^T, bf16 in, fp32 out.
// Workgroup tile 128(M) x 64(N); 8 waves, each 16x64 (1 A-frag, 4 B-frags,
// 4 WMMAs per 32-K step). A/B tiles staged in LDS by the TDM, double-buffered:
// prefetch of buffer p^1 is issued before waiting tensorcnt<=2 for buffer p,
// so the DMA overlaps WMMA compute on the live buffer.
// ---------------------------------------------------------------------------
#define GA_ELEMS (128 * 32)
#define GB_ELEMS (64 * 32)
#define GBUF_ELEMS (GA_ELEMS + GB_ELEMS)

__global__ void __launch_bounds__(256) gemm_bf16_nt(const __bf16* __restrict__ A,
                                                    const __bf16* __restrict__ Bt,
                                                    float* __restrict__ C,
                                                    int N, int Kd) {
  __shared__ __bf16 smem[2][GBUF_ELEMS];   // first (only) LDS object: offset 0
  int wave = threadIdx.x >> 5, lane = threadIdx.x & 31;
  int lane_n = lane & 15, hi = lane >> 4;
  int mblk = blockIdx.y * 128;
  int n0 = blockIdx.x * 64;
  int m0 = mblk + wave * 16;

  // The TDM (not this wave's stores) fills smem; without a visible store LLVM
  // folds every ds_load from smem to undef. This store is never executed
  // (N > 0 always) but is not provably dead, so the loads stay real.
  if (N < 0) smem[0][threadIdx.x] = (__bf16)0.f;

  v8f acc[4];
#pragma unroll
  for (int f = 0; f < 4; ++f) acc[f] = vzero8();

#if HAVE_TDM
  if (wave == 0) {
    tdm_load_2d(A + (size_t)mblk * Kd, 0u, (unsigned)Kd, 1u << 20, 32u, 128u,
                (unsigned)Kd);
    tdm_load_2d(Bt + (size_t)n0 * Kd, (unsigned)(GA_ELEMS * 2), (unsigned)Kd,
                1u << 20, 32u, 64u, (unsigned)Kd);
  }
  int p = 0;
  for (int kk = 0; kk < Kd; kk += 32, p ^= 1) {
    if (wave == 0) {
      if (kk + 32 < Kd) {
        unsigned dst = (unsigned)((p ^ 1) * GBUF_ELEMS * 2);
        tdm_load_2d(A + (size_t)mblk * Kd + kk + 32, dst, (unsigned)Kd, 1u << 20,
                    32u, 128u, (unsigned)Kd);
        tdm_load_2d(Bt + (size_t)n0 * Kd + kk + 32, dst + GA_ELEMS * 2,
                    (unsigned)Kd, 1u << 20, 32u, 64u, (unsigned)Kd);
        __builtin_amdgcn_s_wait_tensorcnt(2);  // buffer p done, prefetch in flight
      } else {
        __builtin_amdgcn_s_wait_tensorcnt(0);
      }
    }
    __syncthreads();  // buffer p visible to all waves
    const __bf16* As = &smem[p][0];
    const __bf16* Bs = &smem[p][GA_ELEMS];
    v16bf a = load_a_frag(As + wave * 16 * 32, 32, lane);
#pragma unroll
    for (int f = 0; f < 4; ++f) {
      v16bf bfr = load_b_frag(Bs + 16 * f * 32, 32, lane);
      acc[f] = wmma_bf16(a, bfr, acc[f]);
    }
    __syncthreads();  // all reads of buffer p done before it is overwritten
  }
#else
  // Fallback: cooperative global->LDS copy (no TDM builtin on this toolchain)
  for (int kk = 0; kk < Kd; kk += 32) {
    int t = threadIdx.x;
#pragma unroll
    for (int r = 0; r < 2; ++r) {  // A tile: 4096 elems = 512 x v8bf
      int v = t + 256 * r;
      int row = (v * 8) >> 5, col = (v * 8) & 31;
      ((v8bf*)&smem[0][0])[v] = *(const v8bf*)(A + (size_t)(mblk + row) * Kd + kk + col);
    }
    {
      int row = (t * 8) >> 5, col = (t * 8) & 31;  // B tile: 2048 elems
      ((v8bf*)&smem[0][GA_ELEMS])[t] = *(const v8bf*)(Bt + (size_t)(n0 + row) * Kd + kk + col);
    }
    __syncthreads();
    const __bf16* As = &smem[0][0];
    const __bf16* Bs = &smem[0][GA_ELEMS];
    v16bf a = load_a_frag(As + wave * 16 * 32, 32, lane);
#pragma unroll
    for (int f = 0; f < 4; ++f) {
      v16bf bfr = load_b_frag(Bs + 16 * f * 32, 32, lane);
      acc[f] = wmma_bf16(a, bfr, acc[f]);
    }
    __syncthreads();
  }
#endif

#pragma unroll
  for (int f = 0; f < 4; ++f)
#pragma unroll
    for (int i = 0; i < 8; ++i)
      C[(size_t)(m0 + i + 8 * hi) * N + n0 + 16 * f + lane_n] = acc[f][i];
}

// ---------------------------------------------------------------------------
// LayerNorm over D (per (b,l) row) + rotary, write bf16 (B,H,L,HD).
// ---------------------------------------------------------------------------
__global__ void __launch_bounds__(256) ln_rope_kernel(const float* __restrict__ y,
                                                      const float* __restrict__ g,
                                                      const float* __restrict__ beta,
                                                      const int* __restrict__ idx,
                                                      __bf16* __restrict__ dst) {
  __shared__ float buf[D_];
  __shared__ float red[16];
  int row = blockIdx.x;                 // b*L + l
  int b = row >> 10;
  int l = row & (L_ - 1);
  int t = threadIdx.x;

  float s = 0.f;
#pragma unroll
  for (int j = 0; j < D_ / 256; ++j) {
    float v = y[(size_t)row * D_ + t + 256 * j];
    buf[t + 256 * j] = v;
    s += v;
  }
#pragma unroll
  for (int off = 16; off; off >>= 1) s += __shfl_xor(s, off, 32);
  if ((t & 31) == 0) red[t >> 5] = s;
  __syncthreads();
  float tot = 0.f;
#pragma unroll
  for (int w = 0; w < 8; ++w) tot += red[w];
  float mu = tot * (1.f / D_);

  float vs = 0.f;
#pragma unroll
  for (int j = 0; j < D_ / 256; ++j) {
    float dv = buf[t + 256 * j] - mu;
    vs += dv * dv;
  }
#pragma unroll
  for (int off = 16; off; off >>= 1) vs += __shfl_xor(vs, off, 32);
  if ((t & 31) == 0) red[8 + (t >> 5)] = vs;
  __syncthreads();
  float tv = 0.f;
#pragma unroll
  for (int w = 0; w < 8; ++w) tv += red[8 + w];
  float rinv = rsqrtf(tv * (1.f / D_) + 1e-5f);

  float pos = (float)idx[row];
#pragma unroll
  for (int j = 0; j < D_ / 256; ++j) {
    int d = t + 256 * j;
    int dh = d & 63, h = d >> 6;
    float xn = (buf[d] - mu) * rinv * g[d] + beta[d];
    int dp = d ^ 32;  // rotary partner within head
    float xp = (buf[dp] - mu) * rinv * g[dp] + beta[dp];
    float fr = rope_freq(dh & 31);
    float ang = pos * fr;
    float rot = (dh < 32) ? -xp : xp;
    float out = xn * __cosf(ang) + rot * __sinf(ang);
    dst[(((size_t)b * H_ + h) * L_ + l) * HD_ + dh] = (__bf16)out;
  }
}

// ---------------------------------------------------------------------------
// V transpose to (B,H,HD,L) bf16 (d-major, so PV B-fragments are contiguous),
// with optional rotary.
// ---------------------------------------------------------------------------
__global__ void __launch_bounds__(256) vtrans_kernel(const float* __restrict__ src,
                                                     const int* __restrict__ idx,
                                                     __bf16* __restrict__ dst,
                                                     int do_rope) {
  int l = blockIdx.x * 256 + threadIdx.x;     // gridDim.x = L/256
  int dlin = blockIdx.y;                      // [0, B*H*HD)
  int b = dlin >> 10;                         // H*HD == 1024
  int hd = dlin & 1023;
  int h = hd >> 6, dh = hd & 63;
  size_t srow = ((size_t)b * L_ + l) * D_;
  float v = src[srow + h * 64 + dh];
  if (do_rope) {
    float vp = src[srow + ((h * 64 + dh) ^ 32)];
    float pos = (float)idx[b * L_ + l];
    float fr = rope_freq(dh & 31);
    float ang = pos * fr;
    float rot = (dh < 32) ? -vp : vp;
    v = v * __cosf(ang) + rot * __sinf(ang);
  }
  dst[(size_t)dlin * L_ + l] = (__bf16)v;
}

// ---------------------------------------------------------------------------
// (B,L,D) fp32 -> (B,H,L,HD) bf16 reshape (cq, ck)
// ---------------------------------------------------------------------------
__global__ void __launch_bounds__(256) reshape_heads_kernel(const float* __restrict__ src,
                                                            __bf16* __restrict__ dst) {
  size_t t = (size_t)blockIdx.x * 256 + threadIdx.x;  // < B*L*D
  int d = (int)(t & (D_ - 1));
  size_t row = t >> 10;
  int b = (int)(row >> 10);
  int l = (int)(row & (L_ - 1));
  dst[(((size_t)b * H_ + (d >> 6)) * L_ + l) * HD_ + (d & 63)] = (__bf16)src[t];
}

// ---------------------------------------------------------------------------
// Fused dual-stream flash attention.
// One wave per 16-query tile; 32-key blocks; online softmax across the
// spm-selected mix of self/cross logits; two value streams accumulated
// separately; inverse rotary on the self stream in the epilogue.
// ---------------------------------------------------------------------------
__global__ void __launch_bounds__(256) flash_kernel(
    const __bf16* __restrict__ Qh, const __bf16* __restrict__ Kh,
    const __bf16* __restrict__ CQh, const __bf16* __restrict__ CKh,
    const __bf16* __restrict__ Vt, const __bf16* __restrict__ CVt,
    const int* __restrict__ z, const unsigned char* __restrict__ mask,
    const int* __restrict__ idx, const int* __restrict__ chain,
    const int* __restrict__ molt, const float* __restrict__ bias,
    __bf16* __restrict__ att) {
  __shared__ __bf16 ldsP[8][2][16 * 32];   // per-wave P_self / P_cross tiles

  int wave = threadIdx.x >> 5, lane = threadIdx.x & 31;
  int lane_n = lane & 15, hi = lane >> 4;
  int qb = (blockIdx.x * 8 + wave) * 16;
  int h = blockIdx.y, b = blockIdx.z;
  const float scale = 0.125f;  // 1/sqrt(HD)

  size_t base = ((size_t)b * H_ + h) * L_ * HD_;
  const __bf16* Q  = Qh  + base;
  const __bf16* K  = Kh  + base;
  const __bf16* CQ = CQh + base;
  const __bf16* CK = CKh + base;
  size_t vbase = ((size_t)b * H_ + h) * HD_ * L_;
  const __bf16* V  = Vt  + vbase;
  const __bf16* CV = CVt + vbase;

  // Q / cQ fragments are loop-invariant: keep resident in VGPRs.
  v16bf qa[2], cqa[2];
#pragma unroll
  for (int c = 0; c < 2; ++c) {
    qa[c]  = load_a_frag(Q  + (size_t)qb * HD_ + 32 * c, HD_, lane);
    cqa[c] = load_a_frag(CQ + (size_t)qb * HD_ + 32 * c, HD_, lane);
  }

  int chq[8];
  float posq[8];
#pragma unroll
  for (int i = 0; i < 8; ++i) {
    int q = qb + i + 8 * hi;
    chq[i]  = chain[b * L_ + q];
    posq[i] = (float)idx[b * L_ + q];
  }

  float m_r[8], l_r[8];
  v8f o_s[4], o_c[4];
#pragma unroll
  for (int i = 0; i < 8; ++i) { m_r[i] = -3.0e38f; l_r[i] = 0.f; }
#pragma unroll
  for (int f = 0; f < 4; ++f) { o_s[f] = vzero8(); o_c[f] = vzero8(); }

  __bf16* Ps = &ldsP[wave][0][0];
  __bf16* Pc = &ldsP[wave][1][0];

  for (int kb = 0; kb < L_; kb += 32) {
    // ---- logits: S_self / S_cross (16 x 32) as two 16x16 C fragments each
    v8f ss[2], sc[2];
#pragma unroll
    for (int j = 0; j < 2; ++j) {
      const __bf16* krow  = K  + (size_t)(kb + 16 * j) * HD_;
      const __bf16* ckrow = CK + (size_t)(kb + 16 * j) * HD_;
      v8f a0 = vzero8(), a1 = vzero8();
#pragma unroll
      for (int c = 0; c < 2; ++c) {
        v16bf kf = load_b_frag(krow + 32 * c, HD_, lane);
        a0 = wmma_bf16(qa[c], kf, a0);
        v16bf ckf = load_b_frag(ckrow + 32 * c, HD_, lane);
        a1 = wmma_bf16(cqa[c], ckf, a1);
      }
      ss[j] = a0;
      sc[j] = a1;
    }

    // ---- spm select + bias gather + mask
    float sval[2][8];
    int issf[2][8];
    float smax[8];
#pragma unroll
    for (int i = 0; i < 8; ++i) smax[i] = -3.0e38f;
#pragma unroll
    for (int j = 0; j < 2; ++j) {
      int k = kb + 16 * j + lane_n;
      int chk = chain[b * L_ + k];
      int mt  = molt[b * L_ + k];
      int mk  = mask[b * L_ + k];
#pragma unroll
      for (int i = 0; i < 8; ++i) {
        int q = qb + i + 8 * hi;
        int sp = (chk == chq[i]) && (mt == 0);
        float v = (sp ? ss[j][i] : sc[j][i]) * scale;
        v += bias[(size_t)z[((size_t)b * L_ + q) * L_ + k] * H_ + h];
        if (!mk) v = -3.0e38f;
        sval[j][i] = v;
        issf[j][i] = sp;
        smax[i] = fmaxf(smax[i], v);
      }
    }

    // ---- online softmax: row reductions across the 16-lane half-wave
#pragma unroll
    for (int i = 0; i < 8; ++i) {
#pragma unroll
      for (int off = 1; off < 16; off <<= 1)
        smax[i] = fmaxf(smax[i], __shfl_xor(smax[i], off, 32));
    }
    float alpha[8], lsum[8];
#pragma unroll
    for (int i = 0; i < 8; ++i) {
      float mn = fmaxf(m_r[i], smax[i]);
      alpha[i] = __expf(m_r[i] - mn);
      m_r[i] = mn;
      lsum[i] = 0.f;
    }
#pragma unroll
    for (int j = 0; j < 2; ++j) {
#pragma unroll
      for (int i = 0; i < 8; ++i) {
        float p = (sval[j][i] <= -1.0e37f) ? 0.f : __expf(sval[j][i] - m_r[i]);
        lsum[i] += p;
        int r = i + 8 * hi, cidx = 16 * j + lane_n;
        Ps[r * 32 + cidx] = (__bf16)(issf[j][i] ? p : 0.f);
        Pc[r * 32 + cidx] = (__bf16)(issf[j][i] ? 0.f : p);
      }
    }
#pragma unroll
    for (int i = 0; i < 8; ++i) {
#pragma unroll
      for (int off = 1; off < 16; off <<= 1)
        lsum[i] += __shfl_xor(lsum[i], off, 32);
      l_r[i] = l_r[i] * alpha[i] + lsum[i];
    }
#pragma unroll
    for (int f = 0; f < 4; ++f)
#pragma unroll
      for (int i = 0; i < 8; ++i) {
        o_s[f][i] *= alpha[i];
        o_c[f][i] *= alpha[i];
      }

    // C-layout -> A-layout via wave-private LDS round trip
    asm volatile("s_wait_dscnt 0" ::: "memory");
    v16bf pa_s = load_a_frag(Ps, 32, lane);
    v16bf pa_c = load_a_frag(Pc, 32, lane);

    // ---- PV: accumulate both value streams (V is d-major: contiguous keys)
#pragma unroll
    for (int f = 0; f < 4; ++f) {
      v16bf vb = load_b_frag(V + (size_t)(16 * f) * L_ + kb, L_, lane);
      o_s[f] = wmma_bf16(pa_s, vb, o_s[f]);
      v16bf cvb = load_b_frag(CV + (size_t)(16 * f) * L_ + kb, L_, lane);
      o_c[f] = wmma_bf16(pa_c, cvb, o_c[f]);
    }
  }

  // ---- epilogue: normalize, inverse rotary on self stream, add cross, store
  float inv[8];
#pragma unroll
  for (int i = 0; i < 8; ++i) inv[i] = (l_r[i] > 0.f) ? 1.f / l_r[i] : 0.f;
#pragma unroll
  for (int c = 0; c < 2; ++c) {
    int d0 = 16 * c + lane_n;             // d0 < 32, pairs with d0+32
    float fr = rope_freq(d0);
#pragma unroll
    for (int i = 0; i < 8; ++i) {
      int q = qb + i + 8 * hi;
      float x0 = o_s[c][i] * inv[i];
      float x1 = o_s[c + 2][i] * inv[i];
      float ang = -posq[i] * fr;
      float cs = __cosf(ang), sn = __sinf(ang);
      float r0 = x0 * cs - x1 * sn;       // rot(d0)   = -x1
      float r1 = x1 * cs + x0 * sn;       // rot(d0+32)=  x0
      float out0 = r0 + o_c[c][i] * inv[i];
      float out1 = r1 + o_c[c + 2][i] * inv[i];
      size_t orow = ((size_t)b * L_ + q) * D_ + h * HD_;
      att[orow + d0]      = (__bf16)out0;
      att[orow + d0 + 32] = (__bf16)out1;
    }
  }
}

// ---------------------------------------------------------------------------
// Host launcher
// ---------------------------------------------------------------------------
extern "C" void kernel_launch(void* const* d_in, const int* in_sizes, int n_in,
                              void* d_out, int out_size, void* d_ws, size_t ws_size,
                              hipStream_t stream) {
  (void)in_sizes; (void)n_in; (void)out_size; (void)ws_size;
  const float* x     = (const float*)d_in[0];
  const int*   z     = (const int*)d_in[1];
  const unsigned char* mask = (const unsigned char*)d_in[2];
  const int*   idx   = (const int*)d_in[3];
  const int*   chain = (const int*)d_in[4];
  const int*   molt  = (const int*)d_in[5];
  const float* W[7]  = { (const float*)d_in[6],   // Wq
                         (const float*)d_in[7],   // Wk
                         (const float*)d_in[8],   // Wv
                         (const float*)d_in[10],  // cWq
                         (const float*)d_in[11],  // cWk
                         (const float*)d_in[12],  // cWv
                         (const float*)d_in[9] }; // Wo (last)
  const float* qg = (const float*)d_in[13];
  const float* qb = (const float*)d_in[14];
  const float* kg = (const float*)d_in[15];
  const float* kb = (const float*)d_in[16];
  const float* bias = (const float*)d_in[17];

  const int M = B_ * L_;                       // 2048
  const size_t MD = (size_t)M * D_;            // 2M elements

  // workspace layout (all regions fully overwritten before use)
  __bf16* xb   = (__bf16*)d_ws;                // M*D bf16
  __bf16* wb   = xb + MD;                      // 7*D*D bf16 (q,k,v,cq,ck,cv,o)
  float*  proj = (float*)(wb + (size_t)7 * D_ * D_);  // 6*M*D fp32
  __bf16* Qh   = (__bf16*)(proj + 6 * MD);     // each M*D bf16
  __bf16* Kh   = Qh  + MD;
  __bf16* CQh  = Kh  + MD;
  __bf16* CKh  = CQh + MD;
  __bf16* Vt   = CKh + MD;
  __bf16* CVt  = Vt  + MD;
  __bf16* att  = CVt + MD;

  // 1) bf16 conversions
  cvt_bf16_kernel<<<(int)(MD / 256), 256, 0, stream>>>(x, xb, (int)MD);
  for (int i = 0; i < 7; ++i)
    cvt_bf16_kernel<<<(D_ * D_) / 256, 256, 0, stream>>>(
        W[i], wb + (size_t)i * D_ * D_, D_ * D_);

  // 2) six projection GEMMs (x @ W.T), WMMA bf16 + TDM-staged LDS tiles
  dim3 ggrid(D_ / 64, M / 128);
  for (int i = 0; i < 6; ++i)
    gemm_bf16_nt<<<ggrid, 256, 0, stream>>>(
        xb, wb + (size_t)i * D_ * D_, proj + (size_t)i * MD, D_, D_);

  // 3) post-processing into attention layouts
  ln_rope_kernel<<<M, 256, 0, stream>>>(proj + 0 * MD, qg, qb, idx, Qh);
  ln_rope_kernel<<<M, 256, 0, stream>>>(proj + 1 * MD, kg, kb, idx, Kh);
  dim3 vgrid(L_ / 256, B_ * H_ * HD_);
  vtrans_kernel<<<vgrid, 256, 0, stream>>>(proj + 2 * MD, idx, Vt, 1);   // v + rope
  vtrans_kernel<<<vgrid, 256, 0, stream>>>(proj + 5 * MD, idx, CVt, 0);  // cv
  reshape_heads_kernel<<<(int)(MD / 256), 256, 0, stream>>>(proj + 3 * MD, CQh);
  reshape_heads_kernel<<<(int)(MD / 256), 256, 0, stream>>>(proj + 4 * MD, CKh);

  // 4) fused dual-stream flash attention
  dim3 fgrid(L_ / (16 * 8), H_, B_);  // 8 waves/WG, one 16-query tile per wave
  flash_kernel<<<fgrid, 256, 0, stream>>>(Qh, Kh, CQh, CKh, Vt, CVt, z, mask, idx,
                                          chain, molt, bias, att);

  // 5) final output GEMM: (out + cross_out) @ Wo.T -> fp32 d_out
  gemm_bf16_nt<<<ggrid, 256, 0, stream>>>(att, wb + (size_t)6 * D_ * D_,
                                          (float*)d_out, D_, D_);
}
